// RoPEMultiheadAttention_2224793059924
// MI455X (gfx1250) — matrix operand order repaired and checked by reading, not compiled, not run on hardware
//
#include <hip/hip_runtime.h>
#include <hip/hip_bf16.h>

// ---------------------------------------------------------------------------
// Problem constants: B=2, L=2048, D=1024, H=16, HD=64
// ---------------------------------------------------------------------------
constexpr int CB = 2, CL = 2048, CD = 1024, CH = 16, CHD = 64;
constexpr int CM = CB * CL; // 4096 rows for the projection GEMMs

typedef __attribute__((ext_vector_type(16))) __bf16 v16bf;
typedef __attribute__((ext_vector_type(8)))  float  v8f;

#define NEG_INF (-__builtin_inff())

__device__ __forceinline__ v8f wmma_bf16(v16bf a, v16bf b, v8f c) {
  return __builtin_amdgcn_wmma_f32_16x16x32_bf16(
      /*neg_a=*/false, a, /*neg_b=*/false, b,
      /*c_mod=*/(short)0, c, /*reuse_a=*/false, /*reuse_b=*/false);
}

// ---------------------------------------------------------------------------
// Intra-16-lane butterfly exchange (lane ^= STEP), VALU-only via
// v_permlane16_b32 (no DS pipe, no DScnt waits) when available.
// ---------------------------------------------------------------------------
template <int STEP>
__device__ __forceinline__ float lane_xor16(float v) {
#if __has_builtin(__builtin_amdgcn_permlane16)
  constexpr unsigned LO = (STEP == 1) ? 0x67452301u
                        : (STEP == 2) ? 0x54761032u
                        : (STEP == 4) ? 0x32107654u
                                      : 0xFEDCBA98u;  // STEP == 8
  constexpr unsigned HI = (STEP == 1) ? 0xEFCDAB89u
                        : (STEP == 2) ? 0xDCFE98BAu
                        : (STEP == 4) ? 0xBA98FEDCu
                                      : 0x76543210u;  // STEP == 8
  unsigned u = __float_as_uint(v);
  unsigned r = __builtin_amdgcn_permlane16(u, u, LO, HI, false, false);
  return __uint_as_float(r);
#else
  return __shfl_xor(v, STEP, 16);
#endif
}

// Batched (8 independent rows) butterfly reductions: 8 exchanges per step in
// flight -> the scheduler overlaps them instead of wait-per-exchange.
__device__ __forceinline__ void rowmax16_b8(float* v) {
#pragma unroll
  for (int i = 0; i < 8; ++i) v[i] = fmaxf(v[i], lane_xor16<1>(v[i]));
#pragma unroll
  for (int i = 0; i < 8; ++i) v[i] = fmaxf(v[i], lane_xor16<2>(v[i]));
#pragma unroll
  for (int i = 0; i < 8; ++i) v[i] = fmaxf(v[i], lane_xor16<4>(v[i]));
#pragma unroll
  for (int i = 0; i < 8; ++i) v[i] = fmaxf(v[i], lane_xor16<8>(v[i]));
}
__device__ __forceinline__ void rowsum16_b8(float* v) {
#pragma unroll
  for (int i = 0; i < 8; ++i) v[i] += lane_xor16<1>(v[i]);
#pragma unroll
  for (int i = 0; i < 8; ++i) v[i] += lane_xor16<2>(v[i]);
#pragma unroll
  for (int i = 0; i < 8; ++i) v[i] += lane_xor16<4>(v[i]);
#pragma unroll
  for (int i = 0; i < 8; ++i) v[i] += lane_xor16<8>(v[i]);
}

// A-operand (16x32 MxK, bf16) fragment loader.  ISA 7.12.2 layout:
// lane<16 : M=lane,    K in {0..7, 16..23}
// lane>=16: M=lane-16, K in {8..15, 24..31}
// rowBase points at (row0, k0) of the tile; strideEl must be even.
__device__ __forceinline__ v16bf load_a_frag(const __bf16* rowBase, int strideEl) {
  int l = threadIdx.x & 31;
  const unsigned* p = (const unsigned*)(rowBase + (size_t)(l & 15) * strideEl);
  int o = (l >> 4) * 4; // uint offset of the K-half
  union { unsigned u[8]; v16bf v; } t;
#pragma unroll
  for (int j = 0; j < 4; ++j) t.u[j] = p[o + j];
#pragma unroll
  for (int j = 0; j < 4; ++j) t.u[4 + j] = p[8 + o + j];
  return t.v;
}

// B-operand (32x16 KxN, bf16) from [k][n] storage: lane -> N=lane&15,
// K in {0..7,16..23} (lane<16) or {8..15,24..31} (lane>=16).
__device__ __forceinline__ v16bf load_row16(const __bf16* p, int strideEl) {
  int l = threadIdx.x & 31;
  const __bf16* c = p + (l & 15);
  int kb = (l >> 4) * 8;
  v16bf t;
#pragma unroll
  for (int s = 0; s < 8; ++s) {
    t[s] = c[(kb + s) * strideEl];
    t[8 + s] = c[(16 + kb + s) * strideEl];
  }
  return t;
}

// ---------------------------------------------------------------------------
// f32 -> bf16 conversion
// ---------------------------------------------------------------------------
__global__ void cvt_bf16_kernel(const float* __restrict__ in,
                                __bf16* __restrict__ out, int n) {
  for (int i = blockIdx.x * blockDim.x + threadIdx.x; i < n;
       i += gridDim.x * blockDim.x)
    out[i] = (__bf16)in[i];
}

// ---------------------------------------------------------------------------
// WMMA GEMM:  out[m,n] = sum_k X[m,k]*W[n,k] + bias[n]   (X:[4096,1024] bf16,
// W:[1024,1024] bf16 row-major over k).  256 threads = 8 waves; block tile
// 128(M) x 128(N); K-step 32.  Wave tile 32x64 = 2x4 WMMA subtiles.
// EPI: 0 = bf16 store permuted to [B,H,L,HD]; 1 = sigmoid -> f32 [m,n];
//      2 = f32 [m,n].
// ---------------------------------------------------------------------------
constexpr int EPI_PERM = 0, EPI_SIG = 1, EPI_LIN = 2;

template <int EPI>
__global__ __launch_bounds__(256) void gemm_wmma(
    const __bf16* __restrict__ X, const __bf16* __restrict__ W,
    const float* __restrict__ bias, void* __restrict__ outp) {
  __shared__ __bf16 ldsA[128 * 40];  // [row][k] stride 40 (pad)
  __shared__ __bf16 ldsB[32 * 136];  // transposed W: [k][n] stride 136 (pad)

  const int blockN = blockIdx.x * 128;
  const int blockM = blockIdx.y * 128;
  const int tid = threadIdx.x;
  const int wave = tid >> 5, l = tid & 31;
  const int wm = wave & 3, wn = wave >> 2;   // 4x2 wave grid
  const int g = l >> 4, nl = l & 15;

  v8f acc[2][4];
#pragma unroll
  for (int i = 0; i < 2; ++i)
#pragma unroll
    for (int j = 0; j < 4; ++j) acc[i][j] = v8f{0.f,0.f,0.f,0.f,0.f,0.f,0.f,0.f};

  for (int k0 = 0; k0 < CD; k0 += 32) {
    // --- stage A tile: 128 rows x 32 cols --------------------------------
    {
      int r = tid >> 1, seg = (tid & 1) * 16;
      const __bf16* src = X + (size_t)(blockM + r) * CD + k0 + seg;
      union { uint4 q[2]; unsigned u[8]; } t;
      t.q[0] = *(const uint4*)src;
      t.q[1] = *(const uint4*)(src + 8);
      unsigned* dst = (unsigned*)&ldsA[r * 40 + seg];
#pragma unroll
      for (int i = 0; i < 8; ++i) dst[i] = t.u[i];
    }
    // --- stage B tile transposed: W rows -> ldsB[k][n] -------------------
    {
      int n = tid >> 1, seg = (tid & 1) * 16;
      const __bf16* src = W + (size_t)(blockN + n) * CD + k0 + seg;
      union { uint4 q[2]; __bf16 e[16]; } t;
      t.q[0] = *(const uint4*)src;
      t.q[1] = *(const uint4*)(src + 8);
#pragma unroll
      for (int i = 0; i < 16; ++i) ldsB[(seg + i) * 136 + n] = t.e[i];
    }
    __syncthreads();

    if (k0 + 32 < CD) {  // hint next K tile (global_prefetch_b8)
      __builtin_prefetch(X + (size_t)(blockM + (tid >> 1)) * CD + k0 + 32, 0, 1);
      __builtin_prefetch(W + (size_t)(blockN + (tid >> 1)) * CD + k0 + 32, 0, 1);
    }

    v16bf a[2], b[4];
    a[0] = load_a_frag(&ldsA[(wm * 32 + 0) * 40], 40);
    a[1] = load_a_frag(&ldsA[(wm * 32 + 16) * 40], 40);
#pragma unroll
    for (int ni = 0; ni < 4; ++ni)
      b[ni] = load_row16(&ldsB[wn * 64 + ni * 16], 136);
#pragma unroll
    for (int mi = 0; mi < 2; ++mi)
#pragma unroll
      for (int ni = 0; ni < 4; ++ni)
        acc[mi][ni] = wmma_bf16(a[mi], b[ni], acc[mi][ni]);
    __syncthreads();
  }

  // --- epilogue ----------------------------------------------------------
#pragma unroll
  for (int mi = 0; mi < 2; ++mi)
#pragma unroll
    for (int ni = 0; ni < 4; ++ni) {
      int nn = blockN + wn * 64 + ni * 16 + nl;
      float bv = bias[nn];
#pragma unroll
      for (int i = 0; i < 8; ++i) {
        int m = blockM + wm * 32 + mi * 16 + i + 8 * g;
        float v = acc[mi][ni][i] + bv;
        if constexpr (EPI == EPI_PERM) {
          int bb = m >> 11, ll = m & (CL - 1);
          int h = nn >> 6, d = nn & (CHD - 1);
          ((__bf16*)outp)[(((size_t)bb * CH + h) * CL + ll) * CHD + d] = (__bf16)v;
        } else if constexpr (EPI == EPI_SIG) {
          ((float*)outp)[(size_t)m * CD + nn] = 1.0f / (1.0f + __expf(-v));
        } else {
          ((float*)outp)[(size_t)m * CD + nn] = v;
        }
      }
    }
}

// ---------------------------------------------------------------------------
// RoPE (in place) on [B,H,L,HD] bf16.  Pairs (d, d+32), d<32.
// ---------------------------------------------------------------------------
__global__ void rope_kernel(__bf16* __restrict__ X, const float* __restrict__ cosb,
                            const float* __restrict__ sinb) {
  int idx = blockIdx.x * blockDim.x + threadIdx.x;  // B*H*L*32 threads
  if (idx >= CB * CH * CL * 32) return;
  int d = idx & 31;
  int ll = (idx >> 5) & (CL - 1);
  int bh = idx >> 16;
  size_t base = ((size_t)bh * CL + ll) * CHD;
  float x1 = (float)X[base + d];
  float x2 = (float)X[base + d + 32];
  float c1 = cosb[ll * CHD + d],      s1 = sinb[ll * CHD + d];
  float c2 = cosb[ll * CHD + d + 32], s2 = sinb[ll * CHD + d + 32];
  X[base + d]      = (__bf16)(x1 * c1 - x2 * s1);
  X[base + d + 32] = (__bf16)(x2 * c2 + x1 * s2);
}

// ---------------------------------------------------------------------------
// Flash attention.  Grid (L/64, B*H); 128 threads = 4 waves; each wave owns
// 16 queries x HD=64.  Key loop in tiles of 32.
// ---------------------------------------------------------------------------
__global__ __launch_bounds__(128) void flash_attn_kernel(
    const __bf16* __restrict__ Qr, const __bf16* __restrict__ Kr,
    const __bf16* __restrict__ Vr, const unsigned char* __restrict__ mask,
    float* __restrict__ outp) {
  __shared__ __bf16 ldsK[64 * 40];      // transposed K tile: [d][key]
  __shared__ __bf16 ldsV[32 * 72];      // V tile: [key][d]
  __shared__ __bf16 ldsP[4 * 16 * 40];  // per-wave P tile [q][key]

  const int bh = blockIdx.y;
  const int bb = bh >> 4;
  const int wave = threadIdx.x >> 5;
  const int l = threadIdx.x & 31;
  const int g = l >> 4, nl = l & 15;
  const int q0 = blockIdx.x * 64 + wave * 16;

  const __bf16* Qbase = Qr + ((size_t)bh * CL + q0) * CHD;
  const v16bf aq0 = load_a_frag(Qbase + 0, CHD);
  const v16bf aq1 = load_a_frag(Qbase + 32, CHD);

  v8f O[4];
  float mrow[8], lrow[8];
#pragma unroll
  for (int n = 0; n < 4; ++n) O[n] = v8f{0.f,0.f,0.f,0.f,0.f,0.f,0.f,0.f};
#pragma unroll
  for (int i = 0; i < 8; ++i) { mrow[i] = NEG_INF; lrow[i] = 0.f; }

  const unsigned char* maskrow = mask + (size_t)bb * CL;
  const float scale = 0.125f;  // 1/sqrt(64)

  for (int kb = 0; kb < CL; kb += 32) {
    // --- stage K (transposed) and V tiles --------------------------------
    {
      int j = threadIdx.x >> 2;            // 0..31 key row
      int seg = (threadIdx.x & 3) * 16;    // 0..48 d segment
      const __bf16* krow = Kr + ((size_t)bh * CL + kb + j) * CHD + seg;
      const __bf16* vrow = Vr + ((size_t)bh * CL + kb + j) * CHD + seg;
      union { uint4 q[2]; __bf16 e[16]; } tk;
      union { uint4 q[2]; unsigned u[8]; } tv;
      tk.q[0] = *(const uint4*)krow; tk.q[1] = *(const uint4*)(krow + 8);
      tv.q[0] = *(const uint4*)vrow; tv.q[1] = *(const uint4*)(vrow + 8);
#pragma unroll
      for (int i = 0; i < 16; ++i) ldsK[(seg + i) * 40 + j] = tk.e[i];
      unsigned* vd = (unsigned*)&ldsV[j * 72 + seg];
#pragma unroll
      for (int i = 0; i < 8; ++i) vd[i] = tv.u[i];
    }
    __syncthreads();

    // --- scores S = Q K^T (16 q x 32 keys), contraction over d=64 --------
    v8f s[2];
    s[0] = v8f{0.f,0.f,0.f,0.f,0.f,0.f,0.f,0.f};
    s[1] = v8f{0.f,0.f,0.f,0.f,0.f,0.f,0.f,0.f};
#pragma unroll
    for (int c = 0; c < 2; ++c) {
      v16bf aq = c ? aq1 : aq0;
#pragma unroll
      for (int t = 0; t < 2; ++t) {
        v16bf bk = load_row16(&ldsK[(c * 32) * 40 + t * 16], 40);
        s[t] = wmma_bf16(aq, bk, s[t]);
      }
    }

    const bool km0 = maskrow[kb + nl] != 0;
    const bool km1 = maskrow[kb + 16 + nl] != 0;

    // --- online softmax (rows striped: q = i + 8*g) ----------------------
    float a0[8], a1[8], mx[8];
#pragma unroll
    for (int i = 0; i < 8; ++i) {
      a0[i] = km0 ? NEG_INF : s[0][i] * scale;
      a1[i] = km1 ? NEG_INF : s[1][i] * scale;
      mx[i] = fmaxf(a0[i], a1[i]);
    }
    rowmax16_b8(mx);  // VALU permlane butterfly, batched for ILP

    float p0[8], p1[8], rs[8];
#pragma unroll
    for (int i = 0; i < 8; ++i) {
      float mnew = fmaxf(mrow[i], mx[i]);
      float alpha, q0v, q1v;
      if (mnew == NEG_INF) {  // row entirely masked so far
        alpha = 1.f; q0v = 0.f; q1v = 0.f;
      } else {
        alpha = (mrow[i] == NEG_INF) ? 0.f : __expf(mrow[i] - mnew);
        q0v = (a0[i] == NEG_INF) ? 0.f : __expf(a0[i] - mnew);
        q1v = (a1[i] == NEG_INF) ? 0.f : __expf(a1[i] - mnew);
      }
      mrow[i] = mnew;
      p0[i] = q0v; p1[i] = q1v;
      rs[i] = q0v + q1v;
      lrow[i] *= alpha;
#pragma unroll
      for (int n = 0; n < 4; ++n) O[n][i] *= alpha;
    }
    rowsum16_b8(rs);

#pragma unroll
    for (int i = 0; i < 8; ++i) {
      lrow[i] += rs[i];
      __bf16* pp = &ldsP[wave * 640 + (i + 8 * g) * 40];
      pp[nl] = (__bf16)p0[i];
      pp[16 + nl] = (__bf16)p1[i];
    }

    // LDS RAW hazard: wave re-reads its own ldsP region (split counters)
    asm volatile("s_wait_dscnt 0" ::: "memory");

    // --- O += P V  (A = P 16x32, B = V 32x16 per d-chunk) ----------------
    v16bf ap = load_a_frag(&ldsP[wave * 640], 40);
#pragma unroll
    for (int n = 0; n < 4; ++n) {
      v16bf bv = load_row16(&ldsV[n * 16], 72);
      O[n] = wmma_bf16(ap, bv, O[n]);
    }
    __syncthreads();
  }

  // --- normalize + store [B,H,L,HD] f32 (all-masked rows -> 0) -----------
#pragma unroll
  for (int n = 0; n < 4; ++n)
#pragma unroll
    for (int i = 0; i < 8; ++i) {
      int q = q0 + i + 8 * g;
      int d = n * 16 + nl;
      float v = (lrow[i] > 0.f) ? (O[n][i] / lrow[i]) : 0.f;
      outp[((size_t)bh * CL + q) * CHD + d] = v;
    }
}

// ---------------------------------------------------------------------------
// X2[m,n] = bf16( attn[b,h,l,d] * gate[m,n] )
// ---------------------------------------------------------------------------
__global__ void gatemul_kernel(const float* __restrict__ attn,
                               const float* __restrict__ gate,
                               __bf16* __restrict__ X2) {
  int idx = blockIdx.x * blockDim.x + threadIdx.x;
  if (idx >= CM * CD) return;
  int m = idx >> 10, n = idx & (CD - 1);
  int bb = m >> 11, ll = m & (CL - 1);
  int h = n >> 6, d = n & (CHD - 1);
  float v = attn[(((size_t)bb * CH + h) * CL + ll) * CHD + d] * gate[idx];
  X2[idx] = (__bf16)v;
}

// ---------------------------------------------------------------------------
// Host-side orchestration
// ---------------------------------------------------------------------------
extern "C" void kernel_launch(void* const* d_in, const int* in_sizes, int n_in,
                              void* d_out, int out_size, void* d_ws, size_t ws_size,
                              hipStream_t stream) {
  const float* query = (const float*)d_in[0];
  const float* key_in = (const float*)d_in[1];
  const float* value = (const float*)d_in[2];
  const unsigned char* mask = (const unsigned char*)d_in[3];
  const float* rope_cos = (const float*)d_in[4];
  const float* rope_sin = (const float*)d_in[5];
  const float* Wq = (const float*)d_in[6];  const float* bq = (const float*)d_in[7];
  const float* Wk = (const float*)d_in[8];  const float* bk = (const float*)d_in[9];
  const float* Wv = (const float*)d_in[10]; const float* bv = (const float*)d_in[11];
  const float* Wg = (const float*)d_in[12]; const float* bg = (const float*)d_in[13];
  const float* Wo = (const float*)d_in[14]; const float* bo = (const float*)d_in[15];

  char* ws = (char*)d_ws;
  size_t off = 0;
  auto alloc = [&](size_t bytes) -> void* {
    void* p = ws + off;
    off += (bytes + 255) & ~(size_t)255;
    return p;
  };
  const size_t ACT = (size_t)CM * CD;       // 4,194,304 elements
  const size_t WEL = (size_t)CD * CD;       // 1,048,576 elements

  __bf16* Xq  = (__bf16*)alloc(ACT * 2);
  __bf16* Xk  = (__bf16*)alloc(ACT * 2);
  __bf16* Xv  = (__bf16*)alloc(ACT * 2);
  __bf16* Wqb = (__bf16*)alloc(WEL * 2);
  __bf16* Wkb = (__bf16*)alloc(WEL * 2);
  __bf16* Wvb = (__bf16*)alloc(WEL * 2);
  __bf16* Wgb = (__bf16*)alloc(WEL * 2);
  __bf16* Wob = (__bf16*)alloc(WEL * 2);
  __bf16* Qr  = (__bf16*)alloc(ACT * 2);    // [B,H,L,HD]
  __bf16* Kr  = (__bf16*)alloc(ACT * 2);
  __bf16* Vr  = (__bf16*)alloc(ACT * 2);
  float*  gateF = (float*)alloc(ACT * 4);   // [m,n]
  float*  attnF = (float*)alloc(ACT * 4);   // [B,H,L,HD]
  __bf16* X2  = (__bf16*)alloc(ACT * 2);
  (void)ws_size; (void)in_sizes; (void)n_in; (void)out_size;

  // 1) convert to bf16
  cvt_bf16_kernel<<<(int)((ACT + 255) / 256), 256, 0, stream>>>(query, Xq, (int)ACT);
  cvt_bf16_kernel<<<(int)((ACT + 255) / 256), 256, 0, stream>>>(key_in, Xk, (int)ACT);
  cvt_bf16_kernel<<<(int)((ACT + 255) / 256), 256, 0, stream>>>(value, Xv, (int)ACT);
  cvt_bf16_kernel<<<(int)((WEL + 255) / 256), 256, 0, stream>>>(Wq, Wqb, (int)WEL);
  cvt_bf16_kernel<<<(int)((WEL + 255) / 256), 256, 0, stream>>>(Wk, Wkb, (int)WEL);
  cvt_bf16_kernel<<<(int)((WEL + 255) / 256), 256, 0, stream>>>(Wv, Wvb, (int)WEL);
  cvt_bf16_kernel<<<(int)((WEL + 255) / 256), 256, 0, stream>>>(Wg, Wgb, (int)WEL);
  cvt_bf16_kernel<<<(int)((WEL + 255) / 256), 256, 0, stream>>>(Wo, Wob, (int)WEL);

  dim3 ggrid(CD / 128, CM / 128);  // (8, 32)
  // 2) projections (store permuted bf16 [B,H,L,HD]) + gate (sigmoid f32)
  gemm_wmma<EPI_PERM><<<ggrid, 256, 0, stream>>>(Xq, Wqb, bq, Qr);
  gemm_wmma<EPI_PERM><<<ggrid, 256, 0, stream>>>(Xk, Wkb, bk, Kr);
  gemm_wmma<EPI_PERM><<<ggrid, 256, 0, stream>>>(Xv, Wvb, bv, Vr);
  gemm_wmma<EPI_SIG ><<<ggrid, 256, 0, stream>>>(Xq, Wgb, bg, gateF);

  // 3) RoPE on Q and K (in place)
  int ropeThreads = CB * CH * CL * 32;
  rope_kernel<<<(ropeThreads + 255) / 256, 256, 0, stream>>>(Qr, rope_cos, rope_sin);
  rope_kernel<<<(ropeThreads + 255) / 256, 256, 0, stream>>>(Kr, rope_cos, rope_sin);

  // 4) flash attention
  flash_attn_kernel<<<dim3(CL / 64, CB * CH), 128, 0, stream>>>(Qr, Kr, Vr, mask, attnF);

  // 5) gate multiply -> bf16
  gatemul_kernel<<<(int)((ACT + 255) / 256), 256, 0, stream>>>(attnF, gateF, X2);

  // 6) output projection -> f32 d_out
  gemm_wmma<EPI_LIN><<<ggrid, 256, 0, stream>>>(X2, Wob, bo, (float*)d_out);
}
